// XNMNet_27092653703936
// MI455X (gfx1250) — matrix operand order repaired and checked by reading, not compiled, not run on hardware
//
#include <hip/hip_runtime.h>
#include <stdint.h>

// CLEVR-scale dims
constexpr int B_ = 128, N_ = 36, DF_ = 512, DE_ = 256, DV_ = 512, H_ = 256, NC_ = 28;
constexpr float SCALE_ = 0.04419417382415922f;   // 1/sqrt(DV)

typedef __attribute__((ext_vector_type(16))) __bf16 v16bf;
typedef __attribute__((ext_vector_type(8)))  float  v8f;

#if __has_builtin(__builtin_amdgcn_cvt_pk_bf16_f32)
#define HAVE_PK_CVT 1
typedef __attribute__((ext_vector_type(2))) __bf16 v2bf;
#endif

// pack two f32 into one dword holding 2 bf16
__device__ __forceinline__ unsigned int pk2bf(float x, float y) {
#ifdef HAVE_PK_CVT
  v2bf r = __builtin_amdgcn_cvt_pk_bf16_f32(x, y);
  return __builtin_bit_cast(unsigned int, r);
#else
  // round-to-nearest (ties away from zero), then pack the two high halves
  // with a single v_perm_b32: dst = {uy[31:24],uy[23:16],ux[31:24],ux[23:16]}
  unsigned int ux = __float_as_uint(x) + 0x8000u;
  unsigned int uy = __float_as_uint(y) + 0x8000u;
  return __builtin_amdgcn_perm(uy, ux, 0x07060302u);
#endif
}

__device__ __forceinline__ unsigned short f2bf(float f) {   // exact RNE (weights)
  unsigned int u = __float_as_uint(f);
  u += 0x7FFFu + ((u >> 16) & 1u);
  return (unsigned short)(u >> 16);
}

// convert 8 consecutive f32 at p into 4 packed-bf16 dwords
__device__ __forceinline__ void load8_bf16(const float* __restrict__ p, unsigned int* d) {
  const float4* p4 = (const float4*)p;
  float4 a = p4[0], b = p4[1];
  d[0] = pk2bf(a.x, a.y);
  d[1] = pk2bf(a.z, a.w);
  d[2] = pk2bf(b.x, b.y);
  d[3] = pk2bf(b.z, b.w);
}

union Frag {
  v16bf v;
  unsigned short s[16];
  unsigned int u[8];
  uint4 q[2];
};

// ---------------------------------------------------------------------------
// Pack a row-major f32 weight matrix W[K x 256] into bf16 WMMA-B fragments.
// Fragment f = nt*kchunks + kc holds B[kc*32 .. kc*32+31, nt*16 .. nt*16+15]:
//   element j of lane L:  B[kc*32 + 8*(L>>4) + (j<8 ? j : j+8), nt*16 + (L&15)]
// stored at out[f*512 + L*16 + j].
// ---------------------------------------------------------------------------
__global__ void pack_w_bf16(const float* __restrict__ W, unsigned short* __restrict__ out,
                            int kchunks, int nfrags) {
  int frag = blockIdx.x * 8 + (threadIdx.x >> 5);
  int lane = threadIdx.x & 31;
  if (frag >= nfrags) return;
  int nt = frag / kchunks, kc = frag % kchunks;
  int n  = nt * 16 + (lane & 15);
  int kb = kc * 32 + 8 * (lane >> 4);
  unsigned short* dst = out + (size_t)frag * 512 + lane * 16;
#pragma unroll
  for (int j = 0; j < 16; ++j)
    dst[j] = f2bf(W[(size_t)(kb + (j < 8 ? j : j + 8)) * 256 + n]);
}

// ---------------------------------------------------------------------------
// q4[b][l][d] = word_emb[program_inputs[b,l]][d]   for l in 0..3
// ---------------------------------------------------------------------------
__global__ void gather_q(const float* __restrict__ word_emb, const int* __restrict__ pin,
                         float* __restrict__ q4) {
  int tid = blockIdx.x * blockDim.x + threadIdx.x;          // B*4*512
  int b = tid >> 11;
  int rem = tid & 2047;
  int l = rem >> 9;
  int d = rem & 511;
  int idx = pin[b * 5 + l];
  q4[tid] = word_emb[(size_t)idx * 512 + d];
}

// ---------------------------------------------------------------------------
// Per-batch projections: v3=W_n2 q3, v1=W_n2 q1, u2=W_e2 q2 (+ bias dots)
// ---------------------------------------------------------------------------
__global__ void proj_q(const float* __restrict__ W_n2, const float* __restrict__ b_n2,
                       const float* __restrict__ W_e2, const float* __restrict__ b_e2,
                       const float* __restrict__ q4,
                       float* __restrict__ v3, float* __restrict__ v1, float* __restrict__ u2,
                       float* __restrict__ c3, float* __restrict__ c1, float* __restrict__ d2) {
  int tid = blockIdx.x * blockDim.x + threadIdx.x;
  const int VEC = 3 * B_ * H_;
  if (tid < VEC) {
    int which = tid / (B_ * H_);
    int rem = tid % (B_ * H_);
    int b = rem / H_, h = rem % H_;
    int l = (which == 0) ? 3 : (which == 1) ? 1 : 2;
    const float* W  = (which == 2) ? W_e2 : W_n2;
    const float* qq = q4 + ((size_t)b * 4 + l) * DV_;
    const float* wr = W + (size_t)h * DV_;
    float s = 0.f;
    for (int d = 0; d < DV_; ++d) s += wr[d] * qq[d];
    ((which == 0) ? v3 : (which == 1) ? v1 : u2)[b * H_ + h] = s;
  } else if (tid < VEC + 3 * B_) {
    int t2 = tid - VEC;
    int which = t2 / B_, b = t2 % B_;
    int l = (which == 0) ? 3 : (which == 1) ? 1 : 2;
    const float* bias = (which == 2) ? b_e2 : b_n2;
    const float* qq = q4 + ((size_t)b * 4 + l) * DV_;
    float s = 0.f;
    for (int d = 0; d < DV_; ++d) s += bias[d] * qq[d];
    ((which == 0) ? c3 : (which == 1) ? c1 : d2)[b] = s;
  }
}

// ---------------------------------------------------------------------------
// hn = relu(node_feats @ W_n1 + b_n1)   [4608,512]@[512,256] -> [4608,256]
// One wave per 16x16 tile. Mtiles=288, Ntiles=16, Kchunks=16.
// ---------------------------------------------------------------------------
__global__ __launch_bounds__(256) void node_gemm(const float* __restrict__ A,
                                                 const unsigned short* __restrict__ Bp,
                                                 const float* __restrict__ bias,
                                                 float* __restrict__ hn) {
  int wave = blockIdx.x * 8 + (threadIdx.x >> 5);
  int lane = threadIdx.x & 31;
  int mt = wave >> 4, nt = wave & 15;
  int hlf = lane >> 4;
  int m = mt * 16 + (lane & 15);
  const float* arow = A + (size_t)m * DF_ + hlf * 8;
  v8f acc = {};
#pragma unroll 4
  for (int kc = 0; kc < 16; ++kc) {
    Frag a, b;
    load8_bf16(arow + kc * 32,      a.u);
    load8_bf16(arow + kc * 32 + 16, a.u + 4);
    const uint4* bp = (const uint4*)(Bp + ((size_t)(nt * 16 + kc) * 32 + lane) * 16);
    b.q[0] = bp[0]; b.q[1] = bp[1];
    acc = __builtin_amdgcn_wmma_f32_16x16x32_bf16(false, a.v, false, b.v,
                                                  (short)0, acc, false, false);
  }
  int n = nt * 16 + (lane & 15);
  float bn = bias[n];
#pragma unroll
  for (int r = 0; r < 8; ++r) {
    int row = mt * 16 + hlf * 8 + r;
    float v = acc[r] + bn;
    hn[(size_t)row * H_ + n] = v > 0.f ? v : 0.f;
  }
}

// ---------------------------------------------------------------------------
// Fused edge pipeline: for each of 165888 edge rows e (length 256),
//   h = relu(e @ W_e1 + b_e1);  z = h . u2[b] + d2[b];  weit = sigmoid(z*scale)
// Packed W_e1 (128 KB of bf16 fragments) is async-copied into LDS once per
// block (global_load_async_to_lds_b128 + s_wait_asynccnt). Each wave handles
// TWO 16-row M-tiles with resident A fragments, sharing every B fragment
// between two WMMAs. 10368 tiles -> 5184 jobs -> 648 blocks x 8 waves.
// ---------------------------------------------------------------------------
__global__ __launch_bounds__(256) void edge_fused(const float* __restrict__ E,
                                                  const unsigned short* __restrict__ Bp,
                                                  const float* __restrict__ b_e1,
                                                  const float* __restrict__ u2,
                                                  const float* __restrict__ d2,
                                                  float* __restrict__ weit) {
  __shared__ uint4 smem[8192];                     // 128 KB: 128 frags x 64 uint4
  int tIdx = threadIdx.x;
  int lane = tIdx & 31;

  // cooperative async stage of packed W_e1 into LDS (each lane moves 16B/op)
  {
    unsigned int lds_base = (unsigned int)(uintptr_t)(&smem[0]);
#pragma unroll 4
    for (int i = 0; i < 32; ++i) {
      int idx = i * 256 + tIdx;                    // uint4 index, 8192 total
      unsigned int lds_addr = lds_base + (unsigned int)idx * 16u;
      const void* gaddr = (const void*)((const char*)Bp + (size_t)idx * 16);
      asm volatile("global_load_async_to_lds_b128 %0, %1, off"
                   :: "v"(lds_addr), "v"(gaddr) : "memory");
    }
    asm volatile("s_wait_asynccnt 0x0" ::: "memory");
    __syncthreads();
  }

  int job = blockIdx.x * 8 + (tIdx >> 5);          // 0..5183
  int mt0 = job * 2, mt1 = job * 2 + 1;
  int hlf = lane >> 4;
  int mrow0 = mt0 * 16 + (lane & 15);
  int mrow1 = mt1 * 16 + (lane & 15);
  int bidx0 = (mt0 * 16) / (N_ * N_);
  int bidx1 = (mt1 * 16) / (N_ * N_);
  const float* arow0 = E + (size_t)mrow0 * DE_ + hlf * 8;
  const float* arow1 = E + (size_t)mrow1 * DE_ + hlf * 8;

  Frag a0[8], a1[8];
#pragma unroll
  for (int kc = 0; kc < 8; ++kc) {
    load8_bf16(arow0 + kc * 32,      a0[kc].u);
    load8_bf16(arow0 + kc * 32 + 16, a0[kc].u + 4);
    load8_bf16(arow1 + kc * 32,      a1[kc].u);
    load8_bf16(arow1 + kc * 32 + 16, a1[kc].u + 4);
  }

  int ncol = lane & 15;
  v8f zacc0 = {}, zacc1 = {};
  for (int nt = 0; nt < 16; ++nt) {
    v8f acc0 = {}, acc1 = {};
#pragma unroll
    for (int kc = 0; kc < 8; ++kc) {
      Frag b;
      const uint4* bp = &smem[(nt * 8 + kc) * 64 + lane * 2];
      b.q[0] = bp[0]; b.q[1] = bp[1];
      acc0 = __builtin_amdgcn_wmma_f32_16x16x32_bf16(false, a0[kc].v, false, b.v,
                                                     (short)0, acc0, false, false);
      acc1 = __builtin_amdgcn_wmma_f32_16x16x32_bf16(false, a1[kc].v, false, b.v,
                                                     (short)0, acc1, false, false);
    }
    int n = nt * 16 + ncol;
    float bn = b_e1[n];
    float un0 = u2[bidx0 * H_ + n];
    float un1 = u2[bidx1 * H_ + n];
#pragma unroll
    for (int r = 0; r < 8; ++r) {
      float h0 = acc0[r] + bn;
      float h1 = acc1[r] + bn;
      zacc0[r] += (h0 > 0.f ? h0 : 0.f) * un0;
      zacc1[r] += (h1 > 0.f ? h1 : 0.f) * un1;
    }
  }
  // reduce across the 16 lanes of each half-wave (column dimension)
#pragma unroll
  for (int r = 0; r < 8; ++r) {
    float z0 = zacc0[r], z1 = zacc1[r];
    z0 += __shfl_xor(z0, 1, 32);  z1 += __shfl_xor(z1, 1, 32);
    z0 += __shfl_xor(z0, 2, 32);  z1 += __shfl_xor(z1, 2, 32);
    z0 += __shfl_xor(z0, 4, 32);  z1 += __shfl_xor(z1, 4, 32);
    z0 += __shfl_xor(z0, 8, 32);  z1 += __shfl_xor(z1, 8, 32);
    zacc0[r] = z0; zacc1[r] = z1;
  }
  if ((lane & 15) == 0) {
    float dd0 = d2[bidx0], dd1 = d2[bidx1];
#pragma unroll
    for (int r = 0; r < 8; ++r) {
      int row0 = mt0 * 16 + hlf * 8 + r;
      int row1 = mt1 * 16 + hlf * 8 + r;
      float z0 = (zacc0[r] + dd0) * SCALE_;
      float z1 = (zacc1[r] + dd1) * SCALE_;
      weit[row0] = 1.0f / (1.0f + __expf(-z0));
      weit[row1] = 1.0f / (1.0f + __expf(-z1));
    }
  }
}

// ---------------------------------------------------------------------------
// sig{3,1}[b,n] = sigmoid((hn[b,n,:] . v{3,1}[b] + c{3,1}[b]) * scale)
// ---------------------------------------------------------------------------
__global__ void node_scores(const float* __restrict__ hn,
                            const float* __restrict__ v3, const float* __restrict__ c3,
                            const float* __restrict__ v1, const float* __restrict__ c1,
                            float* __restrict__ sig3, float* __restrict__ sig1) {
  int tid = blockIdx.x * blockDim.x + threadIdx.x;      // 2 * 4608
  int which = tid / (B_ * N_);
  int rem = tid % (B_ * N_);
  int b = rem / N_;
  const float* v = which ? v1 : v3;
  float s = which ? c1[b] : c3[b];
  const float* h = hn + (size_t)rem * H_;
  const float* vb = v + b * H_;
  for (int k = 0; k < H_; ++k) s += h[k] * vb[k];
  s *= SCALE_;
  (which ? sig1 : sig3)[rem] = 1.0f / (1.0f + __expf(-s));
}

// ---------------------------------------------------------------------------
// attn chain: attn=sig3; relate via weit; clip; *sig1; normalize.
// ---------------------------------------------------------------------------
__global__ void relate_attn(const float* __restrict__ sig3, const float* __restrict__ sig1,
                            const float* __restrict__ weit,
                            float* __restrict__ attn_n, float* __restrict__ tsum) {
  int b = blockIdx.x, t = threadIdx.x;
  __shared__ float a1[N_];
  __shared__ float red[64];
  if (t < N_) a1[t] = sig3[b * N_ + t];
  __syncthreads();
  float a3 = 0.f;
  if (t < N_) {
    float s = 0.f;
    const float* wb = weit + (size_t)b * N_ * N_;
    for (int i = 0; i < N_; ++i) s += a1[i] * wb[i * N_ + t];
    s = fminf(fmaxf(s, 0.f), 1.f);
    a3 = s * sig1[b * N_ + t];
  }
  red[t] = a3;
  __syncthreads();
  for (int off = 32; off > 0; off >>= 1) {
    if (t < off) red[t] += red[t + off];
    __syncthreads();
  }
  float denom = red[0] + 1e-8f;
  if (t < N_) attn_n[b * N_ + t] = a3 / denom;
  if (t == 0) tsum[b] = red[0] / denom;
}

// ph[b,h] = sum_n attn_n[b,n] * hn[b,n,h]
__global__ void pool_h(const float* __restrict__ attn_n, const float* __restrict__ hn,
                       float* __restrict__ ph) {
  int tid = blockIdx.x * blockDim.x + threadIdx.x;      // B*H
  int b = tid / H_, h = tid % H_;
  float s = 0.f;
  const float* an = attn_n + b * N_;
  const float* hb = hn + (size_t)b * N_ * H_ + h;
  for (int n = 0; n < N_; ++n) s += an[n] * hb[(size_t)n * H_];
  ph[tid] = s;
}

// g[b,d] = (ph[b,:] @ W_n2[:,d] + t[b]*b_n2[d]) * q0[b,d]
__global__ void pooled_gate(const float* __restrict__ ph, const float* __restrict__ W_n2,
                            const float* __restrict__ b_n2, const float* __restrict__ tsum,
                            const float* __restrict__ q4, float* __restrict__ g) {
  int tid = blockIdx.x * blockDim.x + threadIdx.x;      // B*DV
  int b = tid / DV_, d = tid % DV_;
  float s = 0.f;
  const float* pb = ph + b * H_;
  for (int h = 0; h < H_; ++h) s += pb[h] * W_n2[(size_t)h * DV_ + d];
  s += tsum[b] * b_n2[d];
  g[tid] = s * q4[((size_t)b * 4 + 0) * DV_ + d];
}

// outv[b,d] = relu(g[b,:] @ W_q[:,d] + b_q[d])
__global__ void query_proj(const float* __restrict__ g, const float* __restrict__ W_q,
                           const float* __restrict__ b_q, float* __restrict__ outv) {
  int tid = blockIdx.x * blockDim.x + threadIdx.x;      // B*DV
  int b = tid / DV_, d = tid % DV_;
  float s = b_q[d];
  const float* gb = g + (size_t)b * DV_;
  for (int e = 0; e < DV_; ++e) s += gb[e] * W_q[(size_t)e * DV_ + d];
  outv[tid] = s > 0.f ? s : 0.f;
}

// hc[b,h] = relu(outv[b,:] @ W_c1[:,h] + b_c1[h])
__global__ void cls_hidden(const float* __restrict__ outv, const float* __restrict__ W_c1,
                           const float* __restrict__ b_c1, float* __restrict__ hc) {
  int tid = blockIdx.x * blockDim.x + threadIdx.x;      // B*H
  int b = tid / H_, h = tid % H_;
  float s = b_c1[h];
  const float* ob = outv + (size_t)b * DV_;
  for (int d = 0; d < DV_; ++d) s += ob[d] * W_c1[(size_t)d * H_ + h];
  hc[tid] = s > 0.f ? s : 0.f;
}

// out[b,c] = hc[b,:] @ W_c2[:,c] + b_c2[c]
__global__ void cls_out(const float* __restrict__ hc, const float* __restrict__ W_c2,
                        const float* __restrict__ b_c2, float* __restrict__ out) {
  int tid = blockIdx.x * blockDim.x + threadIdx.x;      // B*NC
  if (tid >= B_ * NC_) return;
  int b = tid / NC_, c = tid % NC_;
  float s = b_c2[c];
  const float* hb = hc + (size_t)b * H_;
  for (int h = 0; h < H_; ++h) s += hb[h] * W_c2[(size_t)h * NC_ + c];
  out[tid] = s;
}

// ---------------------------------------------------------------------------
extern "C" void kernel_launch(void* const* d_in, const int* in_sizes, int n_in,
                              void* d_out, int out_size, void* d_ws, size_t ws_size,
                              hipStream_t stream) {
  const float* node_feats = (const float*)d_in[0];
  const float* edge_feats = (const float*)d_in[1];
  const float* W_n1 = (const float*)d_in[2];
  const float* b_n1 = (const float*)d_in[3];
  const float* W_n2 = (const float*)d_in[4];
  const float* b_n2 = (const float*)d_in[5];
  const float* W_e1 = (const float*)d_in[6];
  const float* b_e1 = (const float*)d_in[7];
  const float* W_e2 = (const float*)d_in[8];
  const float* b_e2 = (const float*)d_in[9];
  const float* W_q  = (const float*)d_in[10];
  const float* b_q  = (const float*)d_in[11];
  const float* W_c1 = (const float*)d_in[12];
  const float* b_c1 = (const float*)d_in[13];
  const float* W_c2 = (const float*)d_in[14];
  const float* b_c2 = (const float*)d_in[15];
  const float* word_emb = (const float*)d_in[16];
  const int* program_inputs = (const int*)d_in[18];
  float* out = (float*)d_out;

  char* w = (char*)d_ws;
  auto alloc = [&](size_t bytes) -> void* {
    void* p = (void*)w;
    w += (bytes + 255) & ~(size_t)255;
    return p;
  };
  float* q4     = (float*)alloc((size_t)B_ * 4 * DV_ * 4);
  float* v3     = (float*)alloc((size_t)B_ * H_ * 4);
  float* v1     = (float*)alloc((size_t)B_ * H_ * 4);
  float* u2     = (float*)alloc((size_t)B_ * H_ * 4);
  float* c3     = (float*)alloc((size_t)B_ * 4);
  float* c1     = (float*)alloc((size_t)B_ * 4);
  float* d2     = (float*)alloc((size_t)B_ * 4);
  float* hn     = (float*)alloc((size_t)B_ * N_ * H_ * 4);
  float* sig3   = (float*)alloc((size_t)B_ * N_ * 4);
  float* sig1   = (float*)alloc((size_t)B_ * N_ * 4);
  float* weit   = (float*)alloc((size_t)B_ * N_ * N_ * 4);
  float* attn_n = (float*)alloc((size_t)B_ * N_ * 4);
  float* tsum   = (float*)alloc((size_t)B_ * 4);
  float* ph     = (float*)alloc((size_t)B_ * H_ * 4);
  float* g      = (float*)alloc((size_t)B_ * DV_ * 4);
  float* outv   = (float*)alloc((size_t)B_ * DV_ * 4);
  float* hc     = (float*)alloc((size_t)B_ * H_ * 4);
  unsigned short* Wn1p = (unsigned short*)alloc((size_t)DF_ * H_ * 2);
  unsigned short* We1p = (unsigned short*)alloc((size_t)DE_ * H_ * 2);

  // 1) weight packing (once per launch; deterministic)
  pack_w_bf16<<<32, 256, 0, stream>>>(W_n1, Wn1p, 16, 256);   // 512x256
  pack_w_bf16<<<16, 256, 0, stream>>>(W_e1, We1p, 8, 128);    // 256x256

  // 2) query gather + per-batch projections
  gather_q<<<(B_ * 4 * DV_) / 256, 256, 0, stream>>>(word_emb, program_inputs, q4);
  proj_q<<<(3 * B_ * H_ + 3 * B_ + 255) / 256, 256, 0, stream>>>(
      W_n2, b_n2, W_e2, b_e2, q4, v3, v1, u2, c3, c1, d2);

  // 3) WMMA GEMMs
  node_gemm<<<(288 * 16) / 8, 256, 0, stream>>>(node_feats, Wn1p, b_n1, hn);
  edge_fused<<<648, 256, 0, stream>>>(edge_feats, We1p, b_e1, u2, d2, weit);

  // 4) attention chain
  node_scores<<<(2 * B_ * N_) / 256, 256, 0, stream>>>(hn, v3, c3, v1, c1, sig3, sig1);
  relate_attn<<<B_, 64, 0, stream>>>(sig3, sig1, weit, attn_n, tsum);

  // 5) query head + classifier
  pool_h<<<(B_ * H_) / 256, 256, 0, stream>>>(attn_n, hn, ph);
  pooled_gate<<<(B_ * DV_) / 256, 256, 0, stream>>>(ph, W_n2, b_n2, tsum, q4, g);
  query_proj<<<(B_ * DV_) / 256, 256, 0, stream>>>(g, W_q, b_q, outv);
  cls_hidden<<<(B_ * H_) / 256, 256, 0, stream>>>(outv, W_c1, b_c1, hc);
  cls_out<<<(B_ * NC_ + 255) / 256, 256, 0, stream>>>(hc, W_c2, b_c2, out);
}
